// MultiHeadAttention_10505490006584
// MI455X (gfx1250) — compile-verified, matrix operand order checked by reading, not run on hardware
//
#include <hip/hip_runtime.h>
#include <hip/hip_bf16.h>
#include <math.h>

// Problem constants (match reference)
#define BB 4
#define SS 2048
#define HH 16
#define DD 128
#define UU 2048            // H*D
#define MM (BB*SS)         // 8192 flattened rows

typedef __attribute__((ext_vector_type(16))) _Float16     v16h;
typedef __attribute__((ext_vector_type(8)))  float        v8f;
typedef __attribute__((ext_vector_type(4)))  unsigned int u32x4;

union Frag16 { u32x4 u4[2]; v16h v; };
union H8     { u32x4 u;     _Float16 h[8]; };

// CDNA5 16-bit A/B fragment layout (ISA 7.12.2): lane-half hf holds
// K = 8*hf + {0..7} (elems 0..7) and K = 16 + 8*hf + {0..7} (elems 8..15),
// i.e. two contiguous 16-byte runs per lane from a row-major [idx][K] tile.
__device__ __forceinline__ v16h ld_frag(const _Float16* tile, int stride, int idx, int hf) {
  const _Float16* r = tile + idx * stride + hf * 8;
  Frag16 f;
  f.u4[0] = *(const u32x4*)(r);        // K = 8*hf + 0..7
  f.u4[1] = *(const u32x4*)(r + 16);   // K = 16 + 8*hf + 0..7
  return f.v;
}

__device__ __forceinline__ v8f wmma_f16(v16h a, v16h b, v8f c) {
  // (neg_a, A, neg_b, B, c_mod, C, reuse_a, reuse_b)
  return __builtin_amdgcn_wmma_f32_16x16x32_f16(false, a, false, b, (short)0, c, false, false);
}

// ---------------------------------------------------------------- convert
__global__ void cvt_f32_f16(const float* __restrict__ s, _Float16* __restrict__ d, int n) {
  int i = blockIdx.x * blockDim.x + threadIdx.x;
  for (; i < n; i += gridDim.x * blockDim.x) d[i] = (_Float16)s[i];
}

// ---------------------------------------------------------------- QKV projection
// One block: 64 rows of x (LDS, reused 3x) x 64 cols of each of Wq/Wk/Wv.
__global__ __launch_bounds__(128)
void qkv_kernel(const _Float16* __restrict__ xh,
                const _Float16* __restrict__ Wqh, const _Float16* __restrict__ Wkh,
                const _Float16* __restrict__ Wvh,
                const float* __restrict__ bq, const float* __restrict__ bk,
                const float* __restrict__ bv,
                _Float16* __restrict__ Qh, _Float16* __restrict__ Kh, _Float16* __restrict__ Vh)
{
  __shared__ _Float16 Xs[64 * 136];   // [m][k], padded stride breaks bank conflicts
  __shared__ _Float16 Wt[64 * 136];   // W transposed: [n][k]
  const int tid = threadIdx.x, wave = tid >> 5, lane = tid & 31;
  const int ln = lane & 15, hf = lane >> 4;
  const int n0 = blockIdx.x * 64, m0 = blockIdx.y * 64;

  for (int i = tid; i < 1024; i += 128) {            // 64 rows x 16 u32x4
    int row = i >> 4, seg = i & 15;
    *(u32x4*)&Xs[row * 136 + seg * 8] = *(const u32x4*)&xh[(size_t)(m0 + row) * DD + seg * 8];
  }

  for (int w = 0; w < 3; ++w) {
    const _Float16* W   = (w == 0) ? Wqh : ((w == 1) ? Wkh : Wvh);
    const float*   bias = (w == 0) ? bq  : ((w == 1) ? bk  : bv);
    _Float16*      outp = (w == 0) ? Qh  : ((w == 1) ? Kh  : Vh);
    __syncthreads();                                  // guard Wt reuse / Xs readiness
    for (int i = tid; i < 1024; i += 128) {           // 128 k-rows x 8 u32x4, transpose into LDS
      int k = i >> 3, seg = i & 7;
      H8 t; t.u = *(const u32x4*)&W[(size_t)k * UU + n0 + seg * 8];
      #pragma unroll
      for (int j = 0; j < 8; ++j) Wt[(seg * 8 + j) * 136 + k] = t.h[j];
    }
    __syncthreads();

    v8f acc[4];
    for (int nt = 0; nt < 4; ++nt)
      for (int e = 0; e < 8; ++e) acc[nt][e] = 0.f;
    #pragma unroll
    for (int kk = 0; kk < 4; ++kk) {                  // D=128 -> 4 steps of K=32
      v16h a = ld_frag(Xs + (wave * 16) * 136 + kk * 32, 136, ln, hf);
      #pragma unroll
      for (int nt = 0; nt < 4; ++nt) {
        v16h bfr = ld_frag(Wt + (nt * 16) * 136 + kk * 32, 136, ln, hf);
        acc[nt] = wmma_f16(a, bfr, acc[nt]);
      }
    }
    // epilogue: + bias, scatter to [B,H,S,D] f16 head-split layout
    #pragma unroll
    for (int nt = 0; nt < 4; ++nt) {
      int ncol = n0 + nt * 16 + ln;
      int hh = ncol >> 7, dd = ncol & 127;
      float bias_v = bias[ncol];
      #pragma unroll
      for (int r = 0; r < 8; ++r) {
        int rg = m0 + wave * 16 + r + 8 * hf;         // C-frag row mapping
        int bb = rg >> 11, s = rg & 2047;
        outp[(((size_t)bb * HH + hh) * SS + s) * DD + dd] = (_Float16)(acc[nt][r] + bias_v);
      }
    }
  }
}

// ---------------------------------------------------------------- flash attention
// Block = (b,h,q-tile of 64 rows), 4 waves x 16 query rows each. Online softmax.
__global__ __launch_bounds__(128)
void fa_kernel(const _Float16* __restrict__ Qh, const _Float16* __restrict__ Kh,
               const _Float16* __restrict__ Vh, _Float16* __restrict__ Ch)
{
  __shared__ _Float16 Qs[64 * 136];   // [q][d]
  __shared__ _Float16 Ks[64 * 136];   // [key][d]  (d is GEMM-K for Q.K^T -> contiguous B frags)
  __shared__ _Float16 Vt[128 * 72];   // V transposed: [d][key]
  __shared__ _Float16 Ps[64 * 72];    // probabilities [q][key] f16
  const int tid = threadIdx.x, wave = tid >> 5, lane = tid & 31;
  const int ln = lane & 15, hf = lane >> 4;
  const int q0 = blockIdx.x * 64, hh = blockIdx.y, bb = blockIdx.z;
  const _Float16* Qp = Qh + ((size_t)(bb * HH + hh) * SS) * DD;
  const _Float16* Kp = Kh + ((size_t)(bb * HH + hh) * SS) * DD;
  const _Float16* Vp = Vh + ((size_t)(bb * HH + hh) * SS) * DD;
  const float scale = 0.08838834764831845f;  // 1/sqrt(128)

  for (int i = tid; i < 1024; i += 128) {
    int row = i >> 4, seg = i & 15;
    *(u32x4*)&Qs[row * 136 + seg * 8] = *(const u32x4*)&Qp[(size_t)(q0 + row) * DD + seg * 8];
  }

  float mrow[8], lrow[8];
  v8f o[8];                                  // 16 q-rows x 128 d-cols accumulator
  for (int r = 0; r < 8; ++r) { mrow[r] = -INFINITY; lrow[r] = 0.f; }
  for (int t = 0; t < 8; ++t)
    for (int e = 0; e < 8; ++e) o[t][e] = 0.f;

  for (int j0 = 0; j0 < SS; j0 += 64) {
    __syncthreads();                         // guard Ks/Vt overwrite vs prev iter reads
    for (int i = tid; i < 1024; i += 128) {
      int row = i >> 4, seg = i & 15;
      *(u32x4*)&Ks[row * 136 + seg * 8] = *(const u32x4*)&Kp[(size_t)(j0 + row) * DD + seg * 8];
    }
    for (int i = tid; i < 1024; i += 128) {  // V transposed into LDS
      int key = i >> 4, seg = i & 15;
      H8 t; t.u = *(const u32x4*)&Vp[(size_t)(j0 + key) * DD + seg * 8];
      #pragma unroll
      for (int j = 0; j < 8; ++j) Vt[(seg * 8 + j) * 72 + key] = t.h[j];
    }
    __syncthreads();

    // S = Q . K^T  (64 keys per wave-row-block, 4 n-tiles of 16)
    v8f sf[4];
    for (int nt = 0; nt < 4; ++nt)
      for (int e = 0; e < 8; ++e) sf[nt][e] = 0.f;
    #pragma unroll
    for (int kk = 0; kk < 4; ++kk) {
      v16h a = ld_frag(Qs + (wave * 16) * 136 + kk * 32, 136, ln, hf);
      #pragma unroll
      for (int nt = 0; nt < 4; ++nt) {
        v16h bfr = ld_frag(Ks + (nt * 16) * 136 + kk * 32, 136, ln, hf);
        sf[nt] = wmma_f16(a, bfr, sf[nt]);
      }
    }
    #pragma unroll
    for (int nt = 0; nt < 4; ++nt)
      for (int e = 0; e < 8; ++e) sf[nt][e] *= scale;

    // online softmax: row stats live in the 16-lane half-wave owning the row
    float alpha[8];
    #pragma unroll
    for (int r = 0; r < 8; ++r) {
      float mx = fmaxf(fmaxf(sf[0][r], sf[1][r]), fmaxf(sf[2][r], sf[3][r]));
      #pragma unroll
      for (int msk = 1; msk < 16; msk <<= 1) mx = fmaxf(mx, __shfl_xor(mx, msk, 32));
      float mnew = fmaxf(mrow[r], mx);
      alpha[r] = __expf(mrow[r] - mnew);
      mrow[r] = mnew;
      float ps = 0.f;
      #pragma unroll
      for (int nt = 0; nt < 4; ++nt) { float p = __expf(sf[nt][r] - mnew); sf[nt][r] = p; ps += p; }
      #pragma unroll
      for (int msk = 1; msk < 16; msk <<= 1) ps += __shfl_xor(ps, msk, 32);
      lrow[r] = lrow[r] * alpha[r] + ps;
    }
    #pragma unroll
    for (int nt = 0; nt < 4; ++nt)
      #pragma unroll
      for (int r = 0; r < 8; ++r)
        Ps[(wave * 16 + r + 8 * hf) * 72 + nt * 16 + ln] = (_Float16)sf[nt][r];
    #pragma unroll
    for (int t = 0; t < 8; ++t)
      for (int r = 0; r < 8; ++r) o[t][r] *= alpha[r];
    __syncthreads();                         // P visible (dscnt) before A-frag reads

    // O += P . V   (K-dim = 64 keys -> 2 steps of 32)
    #pragma unroll
    for (int kk = 0; kk < 2; ++kk) {
      v16h a = ld_frag(Ps + (wave * 16) * 72 + kk * 32, 72, ln, hf);
      #pragma unroll
      for (int t = 0; t < 8; ++t) {
        v16h bfr = ld_frag(Vt + (t * 16) * 72 + kk * 32, 72, ln, hf);
        o[t] = wmma_f16(a, bfr, o[t]);
      }
    }
  }

  // normalize and write concat layout [B,S,H*D] f16
  #pragma unroll
  for (int t = 0; t < 8; ++t) {
    int dcol = hh * DD + t * 16 + ln;
    #pragma unroll
    for (int r = 0; r < 8; ++r) {
      int s = q0 + wave * 16 + r + 8 * hf;
      Ch[((size_t)bb * SS + s) * UU + dcol] = (_Float16)(o[t][r] / lrow[r]);
    }
  }
}

// ---------------------------------------------------------------- output projection
__global__ __launch_bounds__(128)
void proj_kernel(const _Float16* __restrict__ Ch, const _Float16* __restrict__ Woh,
                 const float* __restrict__ bo, float* __restrict__ out)
{
  __shared__ _Float16 As[64 * 72];   // [m][k-chunk]
  __shared__ _Float16 Bt[64 * 72];   // Wo transposed: [n][k-chunk]
  const int tid = threadIdx.x, wave = tid >> 5, lane = tid & 31;
  const int ln = lane & 15, hf = lane >> 4;
  const int m0 = blockIdx.x * 64, n0 = blockIdx.y * 64;

  v8f acc[4];
  for (int nt = 0; nt < 4; ++nt)
    for (int e = 0; e < 8; ++e) acc[nt][e] = 0.f;

  for (int k0 = 0; k0 < UU; k0 += 64) {
    __syncthreads();
    for (int i = tid; i < 512; i += 128) {
      int row = i >> 3, seg = i & 7;
      *(u32x4*)&As[row * 72 + seg * 8] = *(const u32x4*)&Ch[(size_t)(m0 + row) * UU + k0 + seg * 8];
    }
    for (int i = tid; i < 512; i += 128) {
      int k = i >> 3, seg = i & 7;
      H8 t; t.u = *(const u32x4*)&Woh[(size_t)(k0 + k) * DD + n0 + seg * 8];
      #pragma unroll
      for (int j = 0; j < 8; ++j) Bt[(seg * 8 + j) * 72 + k] = t.h[j];
    }
    __syncthreads();
    #pragma unroll
    for (int kk = 0; kk < 2; ++kk) {
      v16h a = ld_frag(As + (wave * 16) * 72 + kk * 32, 72, ln, hf);
      #pragma unroll
      for (int nt = 0; nt < 4; ++nt) {
        v16h bfr = ld_frag(Bt + (nt * 16) * 72 + kk * 32, 72, ln, hf);
        acc[nt] = wmma_f16(a, bfr, acc[nt]);
      }
    }
  }
  #pragma unroll
  for (int nt = 0; nt < 4; ++nt) {
    int col = n0 + nt * 16 + ln;
    float bias = bo[col];
    #pragma unroll
    for (int r = 0; r < 8; ++r) {
      int row = m0 + wave * 16 + r + 8 * hf;
      out[(size_t)row * DD + col] = acc[nt][r] + bias;
    }
  }
}

// ---------------------------------------------------------------- launcher
extern "C" void kernel_launch(void* const* d_in, const int* in_sizes, int n_in,
                              void* d_out, int out_size, void* d_ws, size_t ws_size,
                              hipStream_t stream) {
  (void)in_sizes; (void)n_in; (void)out_size; (void)ws_size;
  const float* x  = (const float*)d_in[0];
  const float* Wq = (const float*)d_in[1];
  const float* bq = (const float*)d_in[2];
  const float* Wk = (const float*)d_in[3];
  const float* bk = (const float*)d_in[4];
  const float* Wv = (const float*)d_in[5];
  const float* bv = (const float*)d_in[6];
  const float* Wo = (const float*)d_in[7];
  const float* bo = (const float*)d_in[8];
  float* out = (float*)d_out;

  _Float16* ws = (_Float16*)d_ws;
  const size_t nX = (size_t)MM * DD;           // 1,048,576
  const size_t nW = (size_t)DD * UU;           //   262,144
  const size_t nQ = (size_t)BB * HH * SS * DD; // 16,777,216
  size_t off = 0;
  _Float16* xh  = ws + off; off += nX;
  _Float16* Wqh = ws + off; off += nW;
  _Float16* Wkh = ws + off; off += nW;
  _Float16* Wvh = ws + off; off += nW;
  _Float16* Woh = ws + off; off += nW;
  _Float16* Qh  = ws + off; off += nQ;
  _Float16* Kh  = ws + off; off += nQ;
  _Float16* Vh  = ws + off; off += nQ;
  _Float16* Ch  = ws + off; off += nQ;         // total ~132 MB of f16 scratch

  cvt_f32_f16<<<512, 256, 0, stream>>>(x,  xh,  (int)nX);
  cvt_f32_f16<<<256, 256, 0, stream>>>(Wq, Wqh, (int)nW);
  cvt_f32_f16<<<256, 256, 0, stream>>>(Wk, Wkh, (int)nW);
  cvt_f32_f16<<<256, 256, 0, stream>>>(Wv, Wvh, (int)nW);
  cvt_f32_f16<<<256, 256, 0, stream>>>(Wo, Woh, (int)nW);

  qkv_kernel<<<dim3(UU / 64, MM / 64), 128, 0, stream>>>(xh, Wqh, Wkh, Wvh, bq, bk, bv, Qh, Kh, Vh);
  fa_kernel <<<dim3(SS / 64, HH, BB),  128, 0, stream>>>(Qh, Kh, Vh, Ch);
  proj_kernel<<<dim3(MM / 64, DD / 64), 128, 0, stream>>>(Ch, Woh, bo, out);
}
